// AxialAttentionBlockAISummer_26147760898126
// MI455X (gfx1250) — matrix operand order repaired and checked by reading, not compiled, not run on hardware
//
#include <hip/hip_runtime.h>

// ---------------------------------------------------------------------------
// AxialAttentionBlock (AISummer) for MI455X / gfx1250, wave32, WMMA f16 + TDM.
// B=8, C_IN=256, DIM=64, HEADS=8, D_IN=128, DKQ=8, DV=16, QKV=32
// ---------------------------------------------------------------------------

typedef __attribute__((ext_vector_type(16))) _Float16 v16h;
typedef __attribute__((ext_vector_type(8)))  float    v8f;
typedef __attribute__((ext_vector_type(4)))  unsigned v4u;
typedef __attribute__((ext_vector_type(8)))  unsigned v8u;

#define DEVI __device__ __forceinline__

DEVI v8f wmma16(v16h a, v16h b, v8f c) {
  // D = A(16x32 f16) * B(32x16 f16) + C(16x16 f32)
  return __builtin_amdgcn_wmma_f32_16x16x32_f16(false, a, false, b, (short)0, c,
                                                false, false);
}

constexpr int   BB    = 8;
constexpr int   CIN   = 256;
constexpr int   HEADS = 8;
constexpr int   DIN   = 128;
constexpr int   NSEQ  = 512;                // B*DIM sequences per axial pass
constexpr int   NCOL  = NSEQ * 64;          // 32768 GEMM columns
constexpr float EPSF  = 1e-5f;

// ---------------------------------------------------------------------------
// Issue a 2D TDM load: tile_dim0 x tile_dim1 elements of 4 bytes, row stride
// rowStrideElems, global source gaddr -> LDS byte offset ldsOff.
// 2-SGPR-group form (VADDR2/3 = NULL -> up to 2D tensors).  Wave-uniform args.
// ---------------------------------------------------------------------------
DEVI void tdm_load_2d_f32(unsigned ldsOff, const void* gptr,
                          unsigned tile0, unsigned tile1, unsigned rowStrideElems) {
  const unsigned long long ga = (unsigned long long)gptr;
  const v4u g0 = { 1u,                         // count=1, user descriptor
                   ldsOff,                     // lds_addr (bytes)
                   (unsigned)ga,               // global_addr[31:0]
                   (unsigned)((ga >> 32) & 0x1FFFFFFu) | (2u << 30) }; // type=2
  const v8u g1 = { 0x20000u,                   // data_size=4B, no multicast
                   tile0 << 16,                // tensor_dim0[15:0] in [31:16]
                   tile1 << 16,                // tensor_dim1[15:0] in [31:16]
                   tile0 << 16,                // tile_dim0 in [127:112]
                   tile1,                      // tile_dim1 in [143:128]
                   rowStrideElems,             // tensor_dim0_stride[31:0]
                   0u, 0u };                   // stride hi / dim1_stride
  asm volatile("tensor_load_to_lds %0, %1" :: "s"(g0), "s"(g1) : "memory");
}

// ---------------------------------------------------------------------------
// Generic WMMA GEMM:  Y[m,n] = sum_k W[m*K+k] * X[((n/L)*K + k)*L + (n%L)]
//   L = 4096  -> activation is x_in (b, c, 64*64)
//   L = 64    -> activation is sequence layout [seq][c][pos]
//   L = 32768 -> canonical (K x N) row-major
// Block = 128 threads (4 waves); block tile 16(M) x 64(N); wave tile 16x16.
// A (16x32) and B (32x64) chunks staged in LDS via coalesced b128 loads.
// ---------------------------------------------------------------------------
template <int L>
__global__ void k_gemm(const float* __restrict__ W, const float* __restrict__ X,
                       float* __restrict__ Y, int M, int N, int K) {
  __shared__ float sA[16 * 33];   // [m][k], pad 33 (bank spread)
  __shared__ float sB[32 * 66];   // [k][n], pad 66 (half-waves 32 banks apart)
  const int tid = threadIdx.x;
  const int lane = tid & 31, wave = tid >> 5;
  const int hs = lane >> 4, ln = lane & 15;
  const int m0 = blockIdx.x * 16;
  const int n0 = blockIdx.y * 64;          // 64 | L -> tile inside one segment
  const long q = n0 / L, rm0 = n0 % L;
  const int am = tid >> 3;                 // 0..15
  const int ak = (tid & 7) * 4;            // 0..28
  v8f c = {};
  for (int k0 = 0; k0 < K; k0 += 32) {
    __syncthreads();
    {  // stage A: 16x32 floats, one float4 per thread
      if (k0 + 32 < K) __builtin_prefetch(W + (size_t)(m0 + am) * K + k0 + 32 + ak, 0, 1);
      const float4 a4 = *(const float4*)(W + (size_t)(m0 + am) * K + k0 + ak);
      float* d = sA + am * 33 + ak;
      d[0] = a4.x; d[1] = a4.y; d[2] = a4.z; d[3] = a4.w;
    }
#pragma unroll
    for (int r = 0; r < 4; ++r) {  // stage B: 32x64 floats, 4 float4 per thread
      const int v = tid + r * 128;
      const int bk = v >> 4;                 // 0..31
      const int bc = (v & 15) * 4;           // 0..60
      const float4 b4 = *(const float4*)(X + ((size_t)q * K + k0 + bk) * L + rm0 + bc);
      float* d = sB + bk * 66 + bc;
      d[0] = b4.x; d[1] = b4.y; d[2] = b4.z; d[3] = b4.w;
    }
    __syncthreads();
    v16h A, Bv;
#pragma unroll
    for (int j = 0; j < 16; ++j) {
      const int ka = (hs << 4) + j;
      A[j] = (_Float16)sA[ln * 33 + ka];
      const int kb = (hs << 4) + j;
      Bv[j] = (_Float16)sB[kb * 66 + wave * 16 + ln];
    }
    c = wmma16(A, Bv, c);
  }
  const int n = n0 + wave * 16 + ln;
#pragma unroll
  for (int r = 0; r < 8; ++r)
    Y[(size_t)(m0 + r + (hs << 3)) * N + n] = c[r];
}

// ---------------------------------------------------------------------------
// Per-row (channel) mean / rstd over contiguous N elements.
// ---------------------------------------------------------------------------
__global__ void k_rowstats(const float* __restrict__ X, int N,
                           float* __restrict__ mean, float* __restrict__ rstd) {
  const int c = blockIdx.x;
  const float* p = X + (size_t)c * N;
  float s = 0.f, s2 = 0.f;
  for (int i = threadIdx.x; i < N; i += blockDim.x) {
    const float v = p[i];
    s += v; s2 += v * v;
  }
  __shared__ float sh[512];
  sh[threadIdx.x] = s; sh[256 + threadIdx.x] = s2;
  __syncthreads();
  for (int o = 128; o > 0; o >>= 1) {
    if (threadIdx.x < (unsigned)o) {
      sh[threadIdx.x] += sh[threadIdx.x + o];
      sh[256 + threadIdx.x] += sh[256 + threadIdx.x + o];
    }
    __syncthreads();
  }
  if (threadIdx.x == 0) {
    const float m = sh[0] / (float)N;
    const float var = sh[256] / (float)N - m * m;
    mean[c] = m; rstd[c] = rsqrtf(var + EPSF);
  }
}

// ---------------------------------------------------------------------------
// BN + ReLU on conv_in output; re-layout (c, b*4096+h*64+w) -> [b*64+w][c][h]
// ---------------------------------------------------------------------------
__global__ void k_bnin(const float* __restrict__ Y1, const float* __restrict__ g,
                       const float* __restrict__ b, const float* __restrict__ mean,
                       const float* __restrict__ rstd, float* __restrict__ XS) {
  const int e = blockIdx.x * blockDim.x + threadIdx.x;   // 128*32768 elements
  const int c = e >> 15, n = e & 32767;
  const int b0 = n >> 12, hw = n & 4095, hh = hw >> 6, w = hw & 63;
  float v = (Y1[e] - mean[c]) * rstd[c] * g[c] + b[c];
  v = fmaxf(v, 0.f);
  XS[(((size_t)(b0 * 64 + w)) * DIN + c) * 64 + hh] = v;
}

// ---------------------------------------------------------------------------
// Attention scores: dots (WMMA, K=8 zero-padded to 32) + qr/kr (VALU).
// q/k tiles (8x64, row stride 8*NCOL) fetched with the Tensor Data Mover.
// STATS pass: reduce 24-channel sum/sumsq.  APPLY pass: fused BN -> logits.
// One block per (seq, head); 256 threads (8 waves, 2 WMMA tiles each).
// ---------------------------------------------------------------------------
template <bool STATS>
__global__ void k_scores(const float* __restrict__ QKV, const float* __restrict__ rel,
                         const float* __restrict__ ga, const float* __restrict__ ba,
                         const float* __restrict__ mean, const float* __restrict__ rstd,
                         float* __restrict__ LG, float* __restrict__ acc) {
  __shared__ float qf[8 * 64];
  __shared__ float kf[8 * 64];
  __shared__ float sacc[6];
  const int bh = blockIdx.x, b = bh >> 3, h = bh & 7;
  const int tid = threadIdx.x;
  const int lane = tid & 31, wave = tid >> 5;
  const int hs = lane >> 4, ln = lane & 15;
  if (STATS && tid < 6) sacc[tid] = 0.f;
  if (wave == 0) {
    // q rows: o = i*8+h (i=0..7) -> row stride 8*NCOL elements
    tdm_load_2d_f32((unsigned)(size_t)qf, QKV + (size_t)h * NCOL + b * 64,
                    64u, 8u, 8u * NCOL);
    // k rows: o = (8+i)*8+h
    tdm_load_2d_f32((unsigned)(size_t)kf, QKV + (size_t)(64 + h) * NCOL + b * 64,
                    64u, 8u, 8u * NCOL);
    __builtin_amdgcn_s_wait_tensorcnt(0);
  }
  __syncthreads();

  v8f dacc[2];
#pragma unroll
  for (int tt = 0; tt < 2; ++tt) {
    const int t = wave * 2 + tt;
    const int td = (t >> 2) * 16, tj = (t & 3) * 16;
    v16h A, Bv;
#pragma unroll
    for (int j = 0; j < 16; ++j) {
      const int ka = (hs << 4) + j;                            // K index of A
      A[j]  = (ka < 8) ? (_Float16)qf[ka * 64 + td + ln] : (_Float16)0.f;
      const int kb = (hs << 4) + j;                            // K index of B
      Bv[j] = (kb < 8) ? (_Float16)kf[kb * 64 + tj + ln] : (_Float16)0.f;
    }
    v8f c = {};
    dacc[tt] = wmma16(A, Bv, c);
  }

  float ps[6] = {0.f, 0.f, 0.f, 0.f, 0.f, 0.f};
#pragma unroll
  for (int tt = 0; tt < 2; ++tt) {
    const int t = wave * 2 + tt;
    const int td = (t >> 2) * 16, tj = (t & 3) * 16;
    const int j = tj + ln;
#pragma unroll
    for (int r = 0; r < 8; ++r) {
      const int d = td + r + (hs << 3);
      const float dot = dacc[tt][r];
      const int ridx = d - j + 63;
      float qr = 0.f, kr = 0.f;
#pragma unroll
      for (int i = 0; i < 8; ++i) {
        qr += qf[i * 64 + d] * rel[i * 127 + ridx];
        kr += kf[i * 64 + d] * rel[(8 + i) * 127 + ridx];
      }
      if (STATS) {
        ps[0] += qr;  ps[1] += qr * qr;
        ps[2] += kr;  ps[3] += kr * kr;
        ps[4] += dot; ps[5] += dot * dot;
      } else {
        const int c0 = h * 3;
        const float v = (qr  - mean[c0    ]) * rstd[c0    ] * ga[c0    ] + ba[c0    ]
                      + (kr  - mean[c0 + 1]) * rstd[c0 + 1] * ga[c0 + 1] + ba[c0 + 1]
                      + (dot - mean[c0 + 2]) * rstd[c0 + 2] * ga[c0 + 2] + ba[c0 + 2];
        LG[((size_t)bh * 64 + d) * 64 + j] = v;
      }
    }
  }
  if (STATS) {
    __syncthreads();
#pragma unroll
    for (int t = 0; t < 6; ++t) atomicAdd(&sacc[t], ps[t]);
    __syncthreads();
    if (tid < 6) {
      const int ch = h * 3 + (tid >> 1);
      atomicAdd(&acc[(tid & 1) * 24 + ch], sacc[tid]);
    }
  }
}

__global__ void k_zero(float* __restrict__ p, int n) {
  const int i = blockIdx.x * blockDim.x + threadIdx.x;
  if (i < n) p[i] = 0.f;
}

__global__ void k_catfin(const float* __restrict__ acc, float* __restrict__ mean,
                         float* __restrict__ rstd) {
  const int c = threadIdx.x;
  if (c < 24) {
    const float cnt = (float)NSEQ * 64.f * 64.f;   // 2,097,152
    const float m = acc[c] / cnt;
    const float var = acc[24 + c] / cnt - m * m;
    mean[c] = m; rstd[c] = rsqrtf(var + EPSF);
  }
}

// ---------------------------------------------------------------------------
// Softmax over last dim (64); one wave per row, 2 elements per lane.
// ---------------------------------------------------------------------------
__global__ void k_softmax(float* __restrict__ LG) {
  const int row = blockIdx.x * 8 + (threadIdx.x >> 5);   // 262144 rows
  const int lane = threadIdx.x & 31;
  float* p = LG + (size_t)row * 64;
  float a = p[lane], b = p[lane + 32];
  float m = fmaxf(a, b);
#pragma unroll
  for (int o = 16; o > 0; o >>= 1) m = fmaxf(m, __shfl_xor(m, o, 32));
  const float ea = __expf(a - m), eb = __expf(b - m);
  float s = ea + eb;
#pragma unroll
  for (int o = 16; o > 0; o >>= 1) s += __shfl_xor(s, o, 32);
  const float inv = 1.f / s;
  p[lane] = ea * inv; p[lane + 32] = eb * inv;
}

// ---------------------------------------------------------------------------
// sv[b,h,i,d] = sum_j attn[b,h,d,j] * v[b,h,i,j]   (per (b,h): 64x16x64 WMMA)
// SV layout: [(bh)*16 + i][d]
// ---------------------------------------------------------------------------
__global__ void k_sv(const float* __restrict__ AT, const float* __restrict__ QKV,
                     float* __restrict__ SV) {
  const int bh = blockIdx.x, b = bh >> 3, h = bh & 7;
  const int lane = threadIdx.x & 31, wave = threadIdx.x >> 5;
  const int hs = lane >> 4, ln = lane & 15;
  const int td = wave * 16;
  v8f c = {};
  for (int k0 = 0; k0 < 64; k0 += 32) {
    v16h A, Bv;
#pragma unroll
    for (int j = 0; j < 16; ++j) {
      const int ka = k0 + (hs << 4) + j;
      A[j] = (_Float16)AT[((size_t)bh * 64 + td + ln) * 64 + ka];
      const int kb = k0 + (hs << 4) + j;
      Bv[j] = (_Float16)QKV[(size_t)((16 + ln) * 8 + h) * NCOL + b * 64 + kb];
    }
    c = wmma16(A, Bv, c);
  }
#pragma unroll
  for (int r = 0; r < 8; ++r) {
    const int d = td + r + (hs << 3);
    SV[((size_t)bh * 16 + ln) * 64 + d] = c[r];
  }
}

// ---------------------------------------------------------------------------
// sve[b,h,i,d] = sum_j attn[b,h,d,j] * rel_v[i, d-j+63]
// Per fixed d: GEMM M=4096 (bh), N=16 (i), K=64 (j).  One wave per 16x16 tile.
// ---------------------------------------------------------------------------
__global__ void k_sve(const float* __restrict__ AT, const float* __restrict__ rel,
                      float* __restrict__ SVE) {
  const int mt = blockIdx.x;          // 256 tiles of bh
  const int d  = blockIdx.y;          // 64
  const int lane = threadIdx.x;
  const int hs = lane >> 4, ln = lane & 15;
  v8f c = {};
  for (int k0 = 0; k0 < 64; k0 += 32) {
    v16h A, Bv;
#pragma unroll
    for (int j = 0; j < 16; ++j) {
      const int ka = k0 + (hs << 4) + j;
      A[j] = (_Float16)AT[((size_t)(mt * 16 + ln) * 64 + d) * 64 + ka];
      const int kb = k0 + (hs << 4) + j;
      Bv[j] = (_Float16)rel[(16 + ln) * 127 + d - kb + 63];
    }
    c = wmma16(A, Bv, c);
  }
#pragma unroll
  for (int r = 0; r < 8; ++r) {
    const int bh = mt * 16 + r + (hs << 3);
    SVE[((size_t)bh * 16 + ln) * 64 + d] = c[r];
  }
}

// ---------------------------------------------------------------------------
// Out-BN stats over 256 channels (s*128 + h*16 + i), 32768 elems each.
// ---------------------------------------------------------------------------
__global__ void k_outstats(const float* __restrict__ SVE, const float* __restrict__ SV,
                           float* __restrict__ mean, float* __restrict__ rstd) {
  const int c2 = blockIdx.x;
  const int s = c2 >> 7, c = c2 & 127;
  const float* src = s ? SV : SVE;
  float sm = 0.f, s2 = 0.f;
  for (int i = threadIdx.x; i < 32768; i += blockDim.x) {
    const float v = src[((size_t)(i >> 6) * 128 + c) * 64 + (i & 63)];
    sm += v; s2 += v * v;
  }
  __shared__ float sh[512];
  sh[threadIdx.x] = sm; sh[256 + threadIdx.x] = s2;
  __syncthreads();
  for (int o = 128; o > 0; o >>= 1) {
    if (threadIdx.x < (unsigned)o) {
      sh[threadIdx.x] += sh[threadIdx.x + o];
      sh[256 + threadIdx.x] += sh[256 + threadIdx.x + o];
    }
    __syncthreads();
  }
  if (threadIdx.x == 0) {
    const float m = sh[0] / 32768.f;
    const float var = sh[256] / 32768.f - m * m;
    mean[c2] = m; rstd[c2] = rsqrtf(var + EPSF);
  }
}

// ---------------------------------------------------------------------------
// Combine: bn(sve)+bn(sv); re-layout for next axial pass (H<->W swap) or,
// for the FINAL pass, ReLU and write canonical (128 x 32768) conv-out input.
// ---------------------------------------------------------------------------
template <bool FINAL>
__global__ void k_combine(const float* __restrict__ SVE, const float* __restrict__ SV,
                          const float* __restrict__ go, const float* __restrict__ bo,
                          const float* __restrict__ mean, const float* __restrict__ rstd,
                          float* __restrict__ dst) {
  const int e = blockIdx.x * blockDim.x + threadIdx.x;   // 512*128*64 elements
  const int seq = e >> 13, rem = e & 8191;
  const int c = rem >> 6, pos = rem & 63;
  const float v0 = (SVE[e] - mean[c]) * rstd[c] * go[c] + bo[c];
  const int c1 = 128 + c;
  const float v1 = (SV[e] - mean[c1]) * rstd[c1] * go[c1] + bo[c1];
  float v = v0 + v1;
  const int b0 = seq >> 6, u = seq & 63;
  if (FINAL) {
    v = fmaxf(v, 0.f);
    dst[(size_t)c * NCOL + b0 * 4096 + u * 64 + pos] = v;   // (c, b*4096+h*64+w)
  } else {
    dst[(((size_t)(b0 * 64 + pos)) * DIN + c) * 64 + u] = v; // [seq'=b*64+pos][c][u]
  }
}

// ---------------------------------------------------------------------------
// Final: BN(conv_out) + residual(x_in) + ReLU -> d_out (b,256,64,64).
// ---------------------------------------------------------------------------
__global__ void k_final(const float* __restrict__ YO, const float* __restrict__ g,
                        const float* __restrict__ b, const float* __restrict__ mean,
                        const float* __restrict__ rstd, const float* __restrict__ x_in,
                        float* __restrict__ out) {
  const int e = blockIdx.x * blockDim.x + threadIdx.x;    // 256*32768 elements
  const int o = e >> 15, n = e & 32767;
  const int b0 = n >> 12, hw = n & 4095;
  const size_t xi = ((size_t)(b0 * CIN + o) << 12) + hw;
  const float v = (YO[e] - mean[o]) * rstd[o] * g[o] + b[o] + x_in[xi];
  out[xi] = fmaxf(v, 0.f);
}

// ---------------------------------------------------------------------------
extern "C" void kernel_launch(void* const* d_in, const int* in_sizes, int n_in,
                              void* d_out, int out_size, void* d_ws, size_t ws_size,
                              hipStream_t stream) {
  const float* x_in  = (const float*)d_in[0];
  const float* w_in  = (const float*)d_in[1];
  const float* g_in  = (const float*)d_in[2];
  const float* b_in  = (const float*)d_in[3];
  const float* w_out = (const float*)d_in[4];
  const float* g_out = (const float*)d_in[5];
  const float* b_out = (const float*)d_in[6];
  const float* wqkv[2] = {(const float*)d_in[7],  (const float*)d_in[13]};
  const float* relp[2] = {(const float*)d_in[8],  (const float*)d_in[14]};
  const float* gaa [2] = {(const float*)d_in[9],  (const float*)d_in[15]};
  const float* baa [2] = {(const float*)d_in[10], (const float*)d_in[16]};
  const float* goo [2] = {(const float*)d_in[11], (const float*)d_in[17]};
  const float* boo [2] = {(const float*)d_in[12], (const float*)d_in[18]};

  float* ws  = (float*)d_ws;
  float* Y1  = ws;                    // 128*32768   conv_in pre-BN
  float* XS  = Y1  + 4194304;         // 512*128*64  sequence input (H pass)
  float* QK  = XS  + 4194304;         // 256*32768   qkv
  float* LG  = QK  + 8388608;         // 512*8*64*64 logits / attn (in place)
  float* SVE = LG  + 16777216;        // 512*128*64
  float* SVb = SVE + 4194304;         // 512*128*64
  float* X2  = SVb + 4194304;         // 512*128*64  sequence input (W pass)
  float* XF  = X2  + 4194304;         // 128*32768   conv_out input (canonical)
  float* YO  = XF  + 4194304;         // 256*32768   conv_out pre-BN
  float* ST  = YO  + 8388608;         // stats pool
  // ST layout: [0]=mean_in(128) [128]=rstd_in(128) [256]=cat acc(48)
  //            [304]=cat mean(24) [328]=cat rstd(24)
  //            [352]=out mean(256) [608]=out rstd(256)
  //            [864]=mean_out(256) [1120]=rstd_out(256)

  // ---- conv_in GEMM + BN + ReLU + layout for H-axial pass --------------
  k_gemm<4096><<<dim3(8, 512), 128, 0, stream>>>(w_in, x_in, Y1, 128, NCOL, 256);
  k_rowstats<<<128, 256, 0, stream>>>(Y1, NCOL, ST, ST + 128);
  k_bnin<<<16384, 256, 0, stream>>>(Y1, g_in, b_in, ST, ST + 128, XS);

  // ---- two axial attention passes (H then W) ---------------------------
  for (int p = 0; p < 2; ++p) {
    const float* Xp = p ? X2 : XS;
    k_gemm<64><<<dim3(16, 512), 128, 0, stream>>>(wqkv[p], Xp, QK, 256, NCOL, 128);
    k_zero<<<1, 64, 0, stream>>>(ST + 256, 48);
    k_scores<true ><<<4096, 256, 0, stream>>>(QK, relp[p], nullptr, nullptr,
                                              nullptr, nullptr, nullptr, ST + 256);
    k_catfin<<<1, 32, 0, stream>>>(ST + 256, ST + 304, ST + 328);
    k_scores<false><<<4096, 256, 0, stream>>>(QK, relp[p], gaa[p], baa[p],
                                              ST + 304, ST + 328, LG, nullptr);
    k_softmax<<<32768, 256, 0, stream>>>(LG);
    k_sv<<<4096, 128, 0, stream>>>(LG, QK, SVb);
    k_sve<<<dim3(256, 64), 32, 0, stream>>>(LG, relp[p], SVE);
    k_outstats<<<256, 256, 0, stream>>>(SVE, SVb, ST + 352, ST + 608);
    if (p == 0)
      k_combine<false><<<16384, 256, 0, stream>>>(SVE, SVb, goo[p], boo[p],
                                                  ST + 352, ST + 608, X2);
    else
      k_combine<true ><<<16384, 256, 0, stream>>>(SVE, SVb, goo[p], boo[p],
                                                  ST + 352, ST + 608, XF);
  }

  // ---- conv_out GEMM + BN + residual + ReLU ----------------------------
  k_gemm<32768><<<dim3(16, 512), 128, 0, stream>>>(w_out, XF, YO, 256, NCOL, 128);
  k_rowstats<<<256, 256, 0, stream>>>(YO, NCOL, ST + 864, ST + 1120);
  k_final<<<32768, 256, 0, stream>>>(YO, g_out, b_out, ST + 864, ST + 1120,
                                     x_in, (float*)d_out);
}
